// CausalSelfAttention_85796266705179
// MI455X (gfx1250) — matrix operand order repaired and checked by reading, not compiled, not run on hardware
//
#include <hip/hip_runtime.h>
#include <hip/hip_bf16.h>

typedef __bf16 bf16_t;
typedef bf16_t v16bf __attribute__((ext_vector_type(16)));
typedef float  v8f   __attribute__((ext_vector_type(8)));

#define B_    4
#define T_    2048
#define D_    2048
#define H_    16
#define HKV_  4
#define HD_   128
#define ROPE_ 64

// ---------------------------------------------------------------------------
// helpers
// ---------------------------------------------------------------------------
__device__ inline v16bf frag_ab(const bf16_t* p0, const bf16_t* p1) {
  union { v16bf v; uint4 q[2]; } u;
  u.q[0] = *(const uint4*)p0;   // 8 bf16
  u.q[1] = *(const uint4*)p1;   // 8 bf16
  return u.v;
}

__device__ inline v8f wmma_bf16(v16bf a, v16bf b, v8f c) {
  // D(16x16,f32) = A(16x32,bf16) * B(32x16,bf16) + C
  return __builtin_amdgcn_wmma_f32_16x16x32_bf16(
      /*neg_a=*/false, a, /*neg_b=*/false, b,
      /*c_mod=*/(short)0, c, /*reuse_a=*/false, /*reuse_b=*/false);
}

// cache/L2 -> LDS async copy, 16B per lane; tracked by ASYNCcnt (no VGPR data)
__device__ inline void async_copy_b128(const void* gptr, void* lds_ptr) {
  unsigned loff = (unsigned)(unsigned long)lds_ptr;   // low 32 bits = LDS offset
  asm volatile("global_load_async_to_lds_b128 %0, %1, off"
               :: "v"(loff), "v"(gptr)
               : "memory");
}
__device__ inline void wait_asynccnt0() {
  asm volatile("s_wait_asynccnt 0x0" ::: "memory");
}

// ---------------------------------------------------------------------------
// fp32 -> bf16 conversion
// ---------------------------------------------------------------------------
__global__ __launch_bounds__(256) void cvt_f32_bf16(const float* __restrict__ src,
                                                    bf16_t* __restrict__ dst, long n) {
  long i = (long)blockIdx.x * blockDim.x + threadIdx.x;
  long stride = (long)gridDim.x * blockDim.x;
  for (; i < n; i += stride) dst[i] = (bf16_t)src[i];
}

// ---------------------------------------------------------------------------
// GEMM: Out[M,N] = A[M,K] @ W[N,K]^T   (A,W bf16 row-major; Out f32 or bf16)
// block: 256 threads = 8 waves; block tile 64x128; BK=32
// LDS tiles staged with GLOBAL_LOAD_ASYNC_TO_LDS_B128 (ASYNCcnt path)
// ---------------------------------------------------------------------------
#define GBM 64
#define GBN 128
#define GBK 32
#define LPAD 40   // 32 + 8 bf16 -> 80B pitch: 16B aligned, conflict-free

template<int F32OUT>
__global__ __launch_bounds__(256) void gemm_xwT(
    const bf16_t* __restrict__ A, const bf16_t* __restrict__ W,
    float* __restrict__ outF, bf16_t* __restrict__ outB,
    int M, int N, int K)
{
  __shared__ bf16_t As[GBM][LPAD];
  __shared__ bf16_t Ws[GBN][LPAD];

  const int tid  = threadIdx.x;
  const int lane = tid & 31;
  const int wv   = tid >> 5;
  const int wm   = wv & 1;     // 2 waves along M
  const int wn   = wv >> 1;    // 4 waves along N
  const int half = lane >> 4;
  const int l16  = lane & 15;
  const int kb   = half * 8;
  const long bm0 = (long)blockIdx.y * GBM;
  const long bn0 = (long)blockIdx.x * GBN;

  v8f acc[2][2] = {};

  const int arow = tid >> 2, acol = (tid & 3) * 8;
  const int wrow = tid >> 1, wcol = (tid & 1) * 16;
  const bf16_t* aP = &A[(bm0 + arow) * (long)K + acol];
  const bf16_t* wP = &W[(bn0 + wrow) * (long)K + wcol];

  for (int k0 = 0; k0 < K; k0 += GBK) {
    __syncthreads();
    // async cache->LDS staging (no VGPR round trip)
    async_copy_b128(aP + k0,     &As[arow][acol]);
    async_copy_b128(wP + k0,     &Ws[wrow][wcol]);
    async_copy_b128(wP + k0 + 8, &Ws[wrow][wcol + 8]);
    if (k0 + GBK < K) {   // prefetch next k-tile into cache
      __builtin_prefetch(aP + k0 + GBK, 0, 0);
      __builtin_prefetch(wP + k0 + GBK, 0, 0);
    }
    wait_asynccnt0();
    __syncthreads();

    v16bf af[2], bfr[2];
#pragma unroll
    for (int mt = 0; mt < 2; ++mt) {               // A: lane=row M, K groups kb / kb+16
      const bf16_t* p = &As[wm * 32 + mt * 16 + l16][0];
      af[mt] = frag_ab(p + kb, p + kb + 16);
    }
#pragma unroll
    for (int nt = 0; nt < 2; ++nt) {               // B: lane=col N, K groups kb / kb+16
      const bf16_t* p = &Ws[wn * 32 + nt * 16 + l16][0];
      bfr[nt] = frag_ab(p + kb, p + kb + 16);
    }
#pragma unroll
    for (int mt = 0; mt < 2; ++mt)
#pragma unroll
      for (int nt = 0; nt < 2; ++nt)
        acc[mt][nt] = wmma_bf16(af[mt], bfr[nt], acc[mt][nt]);
  }

#pragma unroll
  for (int mt = 0; mt < 2; ++mt)
#pragma unroll
    for (int nt = 0; nt < 2; ++nt)
#pragma unroll
      for (int r = 0; r < 8; ++r) {
        const long m = bm0 + wm * 32 + mt * 16 + half * 8 + r;
        const long n = bn0 + wn * 32 + nt * 16 + l16;
        const float v = acc[mt][nt][r];
        if (F32OUT) outF[m * (long)N + n] = v;
        else        outB[m * (long)N + n] = (bf16_t)v;
      }
}

// ---------------------------------------------------------------------------
// RoPE + gain on q, relayout (B,T,H,hD) -> (B,H,T,hD). Gain folds 1/sqrt(hD).
// ---------------------------------------------------------------------------
__global__ __launch_bounds__(128) void rope_q(
    const bf16_t* __restrict__ q, bf16_t* __restrict__ qr,
    const float* __restrict__ cs, const float* __restrict__ sn,
    const float* __restrict__ gain)
{
  const long row = blockIdx.x;               // (b*T + t)*H + h
  const int  h   = (int)(row % H_);
  const long bt  = row / H_;
  const int  t   = (int)(bt % T_);
  const long b   = bt / T_;
  const int  d   = threadIdx.x;
  const float g  = gain[h] * 0.08838834764831845f;  // q_gain * 1/sqrt(128)

  const bf16_t* src = q + row * HD_;
  bf16_t* dst = qr + (((b * H_ + h) * (long)T_) + t) * HD_;

  float v;
  if (d < ROPE_) {
    const int i = d >> 1;
    const float x1 = (float)src[2 * i], x2 = (float)src[2 * i + 1];
    const float c = cs[(long)t * (ROPE_ / 2) + i];
    const float s = sn[(long)t * (ROPE_ / 2) + i];
    v = (d & 1) ? (x1 * s + x2 * c) : (x1 * c - x2 * s);
  } else {
    v = (float)src[d];
  }
  dst[d] = (bf16_t)(v * g);
}

// RoPE on k; split kv (B,T,Hkv,2*hD) -> k,v in (B,Hkv,T,hD)
__global__ __launch_bounds__(256) void rope_kv(
    const bf16_t* __restrict__ kv, bf16_t* __restrict__ kr, bf16_t* __restrict__ vr,
    const float* __restrict__ cs, const float* __restrict__ sn)
{
  const long row = blockIdx.x;               // (b*T + t)*HKV + hk
  const int  hk  = (int)(row % HKV_);
  const long bt  = row / HKV_;
  const int  t   = (int)(bt % T_);
  const long b   = bt / T_;
  const int  d   = threadIdx.x;              // 0..255

  const bf16_t* src = kv + row * (2 * HD_);
  const long dsti = (((b * HKV_ + hk) * (long)T_) + t) * HD_;

  if (d < HD_) {
    float v;
    if (d < ROPE_) {
      const int i = d >> 1;
      const float x1 = (float)src[2 * i], x2 = (float)src[2 * i + 1];
      const float c = cs[(long)t * (ROPE_ / 2) + i];
      const float s = sn[(long)t * (ROPE_ / 2) + i];
      v = (d & 1) ? (x1 * s + x2 * c) : (x1 * c - x2 * s);
    } else {
      v = (float)src[d];
    }
    kr[dsti + d] = (bf16_t)v;
  } else {
    vr[dsti + (d - HD_)] = src[d];
  }
}

// ---------------------------------------------------------------------------
// Flash-style causal attention. Block = 4 waves = 64 query rows; 32-key tiles.
// K tile staged via async-to-LDS; V tile transposed through VGPRs.
// q,k,v in (B,{H|Hkv},T,hD) bf16; o written back as (B,T,H,hD) bf16.
// ---------------------------------------------------------------------------
#define AQT 64
#define AKT 32
#define KSPITCH 136   // 128+8 bf16 -> 272B pitch
#define VTPITCH 40    // 32+8 bf16 -> 80B pitch
#define PSPITCH 40

__global__ __launch_bounds__(128) void attn_fwd(
    const bf16_t* __restrict__ qr, const bf16_t* __restrict__ kr,
    const bf16_t* __restrict__ vr, bf16_t* __restrict__ ob)
{
  __shared__ bf16_t Ks[AKT][KSPITCH];     // K tile, row-major
  __shared__ bf16_t Vt[HD_][VTPITCH];     // V tile, transposed (d-major)
  __shared__ bf16_t Ps[4][16][PSPITCH];   // per-wave P staging (C->A layout)

  const int  q0  = blockIdx.x * AQT;
  const int  h   = blockIdx.y;
  const long b   = blockIdx.z;
  const int  hk  = h / (H_ / HKV_);
  const int  tid = threadIdx.x, lane = tid & 31, wv = tid >> 5;
  const int  half = lane >> 4, l16 = lane & 15, kb = half * 8;

  const bf16_t* qb  = qr + ((b * H_ + h) * (long)T_) * HD_;
  const bf16_t* kbp = kr + ((b * HKV_ + hk) * (long)T_) * HD_;
  const bf16_t* vbp = vr + ((b * HKV_ + hk) * (long)T_) * HD_;

  // Q fragments live in registers: 16 rows x 128 dims = 4 A-fragments
  v16bf qf[4];
  {
    const bf16_t* qrow = qb + (long)(q0 + wv * 16 + l16) * HD_;
#pragma unroll
    for (int j = 0; j < 4; ++j)
      qf[j] = frag_ab(qrow + j * 32 + kb, qrow + j * 32 + kb + 16);
  }

  v8f oacc[8] = {};
  float mrun[8], lrun[8];
#pragma unroll
  for (int r = 0; r < 8; ++r) { mrun[r] = -1e30f; lrun[r] = 0.f; }

  const int nkt  = (q0 + AQT) / AKT;      // key tiles up to the diagonal
  const int krow = tid >> 2, kcol = (tid & 3) * 32;

  for (int kt = 0; kt < nkt; ++kt) {
    const int kk0 = kt * AKT;
    __syncthreads();
    {
      // K tile: async cache->LDS, 4 x 16B per thread
      const bf16_t* src = kbp + (long)(kk0 + krow) * HD_ + kcol;
#pragma unroll
      for (int j = 0; j < 4; ++j)
        async_copy_b128(src + j * 8, &Ks[krow][kcol + j * 8]);
      // V tile: load + transpose into LDS (overlaps the async copies)
      const bf16_t* vs = vbp + (long)(kk0 + krow) * HD_ + kcol;
#pragma unroll
      for (int j = 0; j < 32; ++j)
        Vt[kcol + j][krow] = vs[j];
    }
    wait_asynccnt0();
    __syncthreads();

    // S(16x32) = Q(16x128) K^T : two 16-key column tiles, 4 K-steps each
    v8f s[2] = {};
#pragma unroll
    for (int nt = 0; nt < 2; ++nt) {
      const bf16_t* kp = &Ks[nt * 16 + l16][0];
#pragma unroll
      for (int ks = 0; ks < 4; ++ks)
        s[nt] = wmma_bf16(qf[ks],
                          frag_ab(kp + ks * 32 + kb, kp + ks * 32 + kb + 16),
                          s[nt]);
    }

    // causal mask + online softmax (per-row reductions over 16-lane groups)
    const int qrow0 = q0 + wv * 16 + half * 8;
#pragma unroll
    for (int r = 0; r < 8; ++r) {
      const int qi = qrow0 + r;
      float s0 = (kk0 + l16      <= qi) ? s[0][r] : -1e30f;
      float s1 = (kk0 + 16 + l16 <= qi) ? s[1][r] : -1e30f;
      float mt = fmaxf(s0, s1);
#pragma unroll
      for (int off = 1; off < 16; off <<= 1)
        mt = fmaxf(mt, __shfl_xor(mt, off, 32));
      const float mnew = fmaxf(mrun[r], mt);
      const float p0 = __expf(s0 - mnew);
      const float p1 = __expf(s1 - mnew);
      float rs = p0 + p1;
#pragma unroll
      for (int off = 1; off < 16; off <<= 1)
        rs += __shfl_xor(rs, off, 32);
      const float corr = __expf(mrun[r] - mnew);
      lrun[r] = lrun[r] * corr + rs;
      mrun[r] = mnew;
#pragma unroll
      for (int nt = 0; nt < 8; ++nt)
        oacc[nt][r] *= corr;
      Ps[wv][half * 8 + r][l16]      = (bf16_t)p0;   // C-layout -> LDS
      Ps[wv][half * 8 + r][16 + l16] = (bf16_t)p1;
    }

    // reload P in A-layout; O(16x128) += P(16x32) V(32x128)
    const bf16_t* pp = &Ps[wv][l16][0];
    const v16bf pf = frag_ab(pp + kb, pp + kb + 16);
#pragma unroll
    for (int nt = 0; nt < 8; ++nt) {
      const bf16_t* vp = &Vt[nt * 16 + l16][0];
      oacc[nt] = wmma_bf16(pf, frag_ab(vp + kb, vp + kb + 16), oacc[nt]);
    }
  }

  // normalize and store o as (B,T,H,hD)
#pragma unroll
  for (int r = 0; r < 8; ++r) {
    const long t = q0 + wv * 16 + half * 8 + r;
    const float inv = 1.0f / lrun[r];
    bf16_t* dst = ob + ((b * T_ + t) * (long)H_ + h) * HD_;
#pragma unroll
    for (int nt = 0; nt < 8; ++nt)
      dst[nt * 16 + l16] = (bf16_t)(oacc[nt][r] * inv);
  }
}

// ---------------------------------------------------------------------------
// launch
// ---------------------------------------------------------------------------
extern "C" void kernel_launch(void* const* d_in, const int* in_sizes, int n_in,
                              void* d_out, int out_size, void* d_ws, size_t ws_size,
                              hipStream_t stream) {
  (void)in_sizes; (void)n_in; (void)out_size; (void)ws_size;

  const float* x    = (const float*)d_in[0];
  const float* cosv = (const float*)d_in[1];
  const float* sinv = (const float*)d_in[2];
  const float* Wq   = (const float*)d_in[3];
  const float* Wkv  = (const float*)d_in[4];
  const float* Wo   = (const float*)d_in[5];
  const float* gain = (const float*)d_in[6];

  const long nX   = (long)B_ * T_ * D_;             // 16,777,216
  const long nWq  = (long)D_ * D_;                  //  4,194,304
  const long nWkv = (long)2 * HKV_ * HD_ * D_;      //  2,097,152
  const long nWo  = (long)D_ * D_;
  const long nKV  = (long)B_ * T_ * 2 * HKV_ * HD_; //  8,388,608
  const long nKr  = (long)B_ * HKV_ * T_ * HD_;     //  4,194,304

  bf16_t* p    = (bf16_t*)d_ws;
  bf16_t* xb   = p; p += nX;
  bf16_t* Wqb  = p; p += nWq;
  bf16_t* Wkvb = p; p += nWkv;
  bf16_t* Wob  = p; p += nWo;
  bf16_t* qbuf = p; p += nX;     // projected q; reused as o after rope_q consumes it
  bf16_t* kvb  = p; p += nKV;
  bf16_t* qrb  = p; p += nX;
  bf16_t* krb  = p; p += nKr;
  bf16_t* vrb  = p; p += nKr;
  bf16_t* obuf = qbuf;

  cvt_f32_bf16<<<4096, 256, 0, stream>>>(x,   xb,   nX);
  cvt_f32_bf16<<<2048, 256, 0, stream>>>(Wq,  Wqb,  nWq);
  cvt_f32_bf16<<<1024, 256, 0, stream>>>(Wkv, Wkvb, nWkv);
  cvt_f32_bf16<<<2048, 256, 0, stream>>>(Wo,  Wob,  nWo);

  // q = x @ Wq^T   (M=8192, N=2048, K=2048) -> (B,T,H,hD) bf16
  gemm_xwT<0><<<dim3(D_ / GBN, (B_ * T_) / GBM), 256, 0, stream>>>(
      xb, Wqb, nullptr, qbuf, B_ * T_, D_, D_);
  // kv = x @ Wkv^T (M=8192, N=1024, K=2048)
  gemm_xwT<0><<<dim3((2 * HKV_ * HD_) / GBN, (B_ * T_) / GBM), 256, 0, stream>>>(
      xb, Wkvb, nullptr, kvb, B_ * T_, 2 * HKV_ * HD_, D_);

  rope_q <<<(unsigned)(B_ * T_ * H_),   128, 0, stream>>>(qbuf, qrb, cosv, sinv, gain);
  rope_kv<<<(unsigned)(B_ * T_ * HKV_), 256, 0, stream>>>(kvb, krb, vrb, cosv, sinv);

  attn_fwd<<<dim3(T_ / AQT, H_, B_), 128, 0, stream>>>(qrb, krb, vrb, obuf);

  // out = o @ Wo^T -> fp32 d_out
  gemm_xwT<1><<<dim3(D_ / GBN, (B_ * T_) / GBM), 256, 0, stream>>>(
      obuf, Wob, (float*)d_out, nullptr, B_ * T_, D_, D_);
}
